// MultiHeadAttention_26800595927538
// MI455X (gfx1250) — matrix operand order repaired and checked by reading, not compiled, not run on hardware
//
#include <hip/hip_runtime.h>

typedef __attribute__((ext_vector_type(16))) __bf16 v16bf;
typedef __attribute__((ext_vector_type(8)))  float  v8f;
typedef __attribute__((ext_vector_type(4)))  unsigned int v4u;

union AF { v16bf v; v4u q[2]; };

constexpr int Bc = 2, Sc = 2048, Dc = 1024, Hc = 16, HDc = 64;

// Low 32 bits of a generic pointer to LDS = LDS byte offset (aperture in high bits).
__device__ inline unsigned lds_off(const void* p) {
  return (unsigned)(size_t)p;
}

// CDNA5 async global->LDS copy, 16 bytes per lane (tracked by ASYNCcnt).
__device__ inline void async_copy_b128(unsigned ldsoff, const void* gptr) {
  asm volatile("global_load_async_to_lds_b128 %0, %1, off"
               :: "v"(ldsoff), "v"(gptr)
               : "memory");
}

__device__ inline void wait_async0() {
  asm volatile("s_wait_asynccnt 0x0" ::: "memory");
}

// ---------------- pack / convert kernels ----------------

__global__ __launch_bounds__(256) void pack_x_kernel(
    const float* __restrict__ q, const float* __restrict__ k,
    const float* __restrict__ v, __bf16* __restrict__ X)
{
  size_t i = (size_t)blockIdx.x * 256 + threadIdx.x;
  const size_t total = (size_t)(Bc * Sc) * (3 * Dc);
  if (i >= total) return;
  const int m = (int)(i / (3 * Dc));
  const int c = (int)(i % (3 * Dc));
  const float* src = (c < Dc) ? q : (c < 2 * Dc) ? k : v;
  const int cc = c & (Dc - 1);
  X[i] = (__bf16)src[(size_t)m * Dc + cc];
}

__global__ __launch_bounds__(256) void cvt_kernel(
    const float* __restrict__ s, __bf16* __restrict__ d, int n)
{
  const int i = blockIdx.x * 256 + threadIdx.x;
  if (i < n) d[i] = (__bf16)s[i];
}

// ---------------- bf16 WMMA GEMM: C = A * B^T + bias ----------------
// A: [M,K] bf16 row-major, Bw: [N,K] bf16 row-major (torch Linear weight).
// 128x128 workgroup tile, 4 waves in a 2x2 grid, 64x64 tile per wave:
// 16 WMMA per 32-wide K-step per wave. Double-buffered LDS tiles filled
// with async global->LDS copies (ASYNCcnt), one barrier per K-step.
// mode 0: scatter into Q[B,H,S,64], K[B,H,S,64], V^T[B,H,64,S] bf16 buffers
// mode 1: write fp32 out[m*N + n] (output projection)

__global__ __launch_bounds__(128) void gemm_kernel(
    const __bf16* __restrict__ A, const __bf16* __restrict__ Bw,
    const float* __restrict__ bias, int M, int N, int K, int mode,
    __bf16* __restrict__ Qh, __bf16* __restrict__ Kh, __bf16* __restrict__ Vt,
    float* __restrict__ outF)
{
  __shared__ __bf16 Als[2][128 * 40];
  __shared__ __bf16 Bls[2][128 * 40];
  const int t = threadIdx.x;
  const int wave = t >> 5, lane = t & 31;
  const int wm = wave >> 1, wn = wave & 1;       // 2x2 wave grid
  const int ml = lane & 15, hl = lane >> 4;
  const int mBase = blockIdx.y * 128;
  const int nBase = blockIdx.x * 128;

  v8f acc[4][4] = {};

  // cooperative staging: thread t owns full 32-wide K row t of both tiles
  const __bf16* gA = A + (size_t)(mBase + t) * K;
  const __bf16* gB = Bw + (size_t)(nBase + t) * K;
  const unsigned aOff = lds_off(&Als[0][t * 40]);
  const unsigned bOff = lds_off(&Bls[0][t * 40]);
  const unsigned bufStride = (unsigned)(128 * 40 * 2);  // bytes per buffer

  const int nk = K >> 5;

  // prologue: async-fill buffer 0 with tile 0 (64 B per thread per matrix)
#pragma unroll
  for (int c = 0; c < 4; ++c) {
    async_copy_b128(aOff + 16 * c, gA + 8 * c);
    async_copy_b128(bOff + 16 * c, gB + 8 * c);
  }
  const __bf16* gan = gA + 32;
  const __bf16* gbn = gB + 32;

  for (int kt = 0; kt < nk; ++kt) {
    wait_async0();       // this wave's copies into buf[kt&1] have landed
    __syncthreads();     // all waves' copies landed; all done reading buf[kt&1^1]

    if (kt + 1 < nk) {   // async-fill the other buffer with the next tile
      const unsigned bsel = (unsigned)((kt + 1) & 1) * bufStride;
#pragma unroll
      for (int c = 0; c < 4; ++c) {
        async_copy_b128(aOff + bsel + 16 * c, gan + 8 * c);
        async_copy_b128(bOff + bsel + 16 * c, gbn + 8 * c);
      }
      gan += 32;
      gbn += 32;
    }

    const __bf16* Ab = Als[kt & 1];
    const __bf16* Bb = Bls[kt & 1];
    AF af[4]; AF bfr[4];
#pragma unroll
    for (int i = 0; i < 4; ++i) {
      const __bf16* p = &Ab[(wm * 64 + i * 16 + ml) * 40];
      af[i].q[0] = *(const v4u*)(p + 16 * hl);       // K = 16h..16h+7
      af[i].q[1] = *(const v4u*)(p + 16 * hl + 8);   // K = 16h+8..16h+15
    }
#pragma unroll
    for (int j = 0; j < 4; ++j) {
      const __bf16* p = &Bb[(wn * 64 + j * 16 + ml) * 40 + 16 * hl];
      bfr[j].q[0] = *(const v4u*)(p);                // K = 16h..16h+15
      bfr[j].q[1] = *(const v4u*)(p + 8);
    }
#pragma unroll
    for (int i = 0; i < 4; ++i)
#pragma unroll
      for (int j = 0; j < 4; ++j)
        acc[i][j] = __builtin_amdgcn_wmma_f32_16x16x32_bf16(
            false, af[i].v, false, bfr[j].v, (short)0, acc[i][j], false, false);
  }

  // epilogue; C layout: row = vv + 8*hl, col = ml
#pragma unroll
  for (int j = 0; j < 4; ++j) {
    const int n = nBase + wn * 64 + j * 16 + ml;
    const float bv = bias[n];
#pragma unroll
    for (int i = 0; i < 4; ++i) {
      const int mT = mBase + wm * 64 + i * 16 + 8 * hl;  // first of 8 rows
      if (mode == 0) {
        const int bb = mT >> 11, s = mT & (Sc - 1);
        const int seg = n >> 10, d = n & (Dc - 1);
        const int hh = d >> 6, hd = d & 63;
        if (seg == 2) {
          // V^T [B,H,64,S]: lane's 8 rows are 8 consecutive s -> one b128
          union { v4u qv; __bf16 e[8]; } pk;
#pragma unroll
          for (int vv = 0; vv < 8; ++vv) pk.e[vv] = (__bf16)(acc[i][j][vv] + bv);
          *(v4u*)&Vt[((size_t)((bb * Hc + hh) * HDc + hd)) * Sc + s] = pk.qv;
        } else {
          __bf16* dst = (seg == 0) ? Qh : Kh;
#pragma unroll
          for (int vv = 0; vv < 8; ++vv)
            dst[((size_t)((bb * Hc + hh) * Sc + (s + vv))) * HDc + hd] =
                (__bf16)(acc[i][j][vv] + bv);
        }
      } else {
#pragma unroll
        for (int vv = 0; vv < 8; ++vv)
          outF[(size_t)(mT + vv) * N + n] = acc[i][j][vv] + bv;
      }
    }
  }
}

// ---------------- flash attention (bf16 WMMA, online softmax) ----------------
// grid: (S/64, B*H), block 128; wave w handles 16 q rows. No barriers in the
// KV loop: K and V^T operands stream from global (contiguous B-fragments),
// the P bounce uses a per-wave LDS tile ordered by DScnt within the wave.

__global__ __launch_bounds__(128) void attn_kernel(
    const __bf16* __restrict__ Qh, const __bf16* __restrict__ Kh,
    const __bf16* __restrict__ Vt, __bf16* __restrict__ O)
{
  __shared__ __bf16 Pl[4][16 * 40];   // per-wave P tile (16x32, pad to 40)
  const int bh = blockIdx.y;
  const int b = bh >> 4, h = bh & 15;
  const int w = threadIdx.x >> 5, lane = threadIdx.x & 31;
  const int ml = lane & 15, hl = lane >> 4;
  const __bf16* Qp = Qh + (size_t)bh * Sc * HDc;
  const __bf16* Kp = Kh + (size_t)bh * Sc * HDc;
  const __bf16* Vp = Vt + (size_t)bh * Sc * HDc;   // [64][S] layout
  const int q0 = blockIdx.x * 64 + w * 16;

  // Q A-fragments for d-chunks [0,32) and [32,64)
  AF qa[2];
  {
    const __bf16* p = Qp + (size_t)(q0 + ml) * HDc;
    qa[0].q[0] = *(const v4u*)(p + 16 * hl);
    qa[0].q[1] = *(const v4u*)(p + 16 * hl + 8);
    qa[1].q[0] = *(const v4u*)(p + 32 + 16 * hl);
    qa[1].q[1] = *(const v4u*)(p + 32 + 16 * hl + 8);
  }

  float mst[8], lst[8];
  v8f o[4] = {};
#pragma unroll
  for (int j = 0; j < 8; ++j) { mst[j] = -3.0e38f; lst[j] = 0.f; }
  const float invs = 0.125f;   // 1/sqrt(64)

  for (int kv = 0; kv < Sc; kv += 32) {
    // scores S = Q K^T : two 16-col tiles, accumulate over d-chunks
    v8f s0 = {}, s1 = {};
#pragma unroll
    for (int c = 0; c < 2; ++c) {
      AF kb0, kb1;
      const __bf16* kp0 = Kp + (size_t)(kv + ml) * HDc + c * 32 + 16 * hl;
      kb0.q[0] = *(const v4u*)(kp0);
      kb0.q[1] = *(const v4u*)(kp0 + 8);
      const __bf16* kp1 = kp0 + 16 * HDc;
      kb1.q[0] = *(const v4u*)(kp1);
      kb1.q[1] = *(const v4u*)(kp1 + 8);
      s0 = __builtin_amdgcn_wmma_f32_16x16x32_bf16(false, qa[c].v, false, kb0.v, (short)0, s0, false, false);
      s1 = __builtin_amdgcn_wmma_f32_16x16x32_bf16(false, qa[c].v, false, kb1.v, (short)0, s1, false, false);
    }
    // online softmax; row m = j + 8*hl spans the 16 lanes of this half-wave
#pragma unroll
    for (int j = 0; j < 8; ++j) {
      const float x0 = s0[j] * invs;
      const float x1 = s1[j] * invs;
      float tmax = fmaxf(x0, x1);
#pragma unroll
      for (int msk = 1; msk < 16; msk <<= 1)
        tmax = fmaxf(tmax, __shfl_xor(tmax, msk, 32));
      const float mnew = fmaxf(mst[j], tmax);
      const float corr = __expf(mst[j] - mnew);
      const float p0 = __expf(x0 - mnew);
      const float p1 = __expf(x1 - mnew);
      float ps = p0 + p1;
#pragma unroll
      for (int msk = 1; msk < 16; msk <<= 1)
        ps += __shfl_xor(ps, msk, 32);
      lst[j] = lst[j] * corr + ps;
      mst[j] = mnew;
#pragma unroll
      for (int dt = 0; dt < 4; ++dt) o[dt][j] *= corr;
      Pl[w][(j + 8 * hl) * 40 + ml]      = (__bf16)p0;
      Pl[w][(j + 8 * hl) * 40 + 16 + ml] = (__bf16)p1;
    }
    // O += P * V  (P from per-wave LDS bounce, V^T streamed from global)
    AF pa;
    {
      const __bf16* p = &Pl[w][ml * 40];
      pa.q[0] = *(const v4u*)(p + 16 * hl);
      pa.q[1] = *(const v4u*)(p + 16 * hl + 8);
    }
#pragma unroll
    for (int dt = 0; dt < 4; ++dt) {
      AF vb;
      const __bf16* p = Vp + (size_t)(dt * 16 + ml) * Sc + kv + 16 * hl;
      vb.q[0] = *(const v4u*)(p);
      vb.q[1] = *(const v4u*)(p + 8);
      o[dt] = __builtin_amdgcn_wmma_f32_16x16x32_bf16(false, pa.v, false, vb.v, (short)0, o[dt], false, false);
    }
  }

  // normalize and store combined-heads layout [B,S,D] bf16
#pragma unroll
  for (int j = 0; j < 8; ++j) lst[j] = 1.0f / lst[j];
#pragma unroll
  for (int dt = 0; dt < 4; ++dt) {
#pragma unroll
    for (int j = 0; j < 8; ++j) {
      const int srow = q0 + j + 8 * hl;
      const int col = h * HDc + dt * 16 + ml;
      O[((size_t)(b * Sc + srow)) * Dc + col] = (__bf16)(o[dt][j] * lst[j]);
    }
  }
}

// ---------------- launch ----------------

extern "C" void kernel_launch(void* const* d_in, const int* in_sizes, int n_in,
                              void* d_out, int out_size, void* d_ws, size_t ws_size,
                              hipStream_t stream) {
  (void)in_sizes; (void)n_in; (void)out_size; (void)ws_size;
  const float* q    = (const float*)d_in[0];
  const float* kk   = (const float*)d_in[1];
  const float* vv   = (const float*)d_in[2];
  const float* Wqkv = (const float*)d_in[3];
  const float* bqkv = (const float*)d_in[4];
  const float* Wo   = (const float*)d_in[5];
  const float* bo   = (const float*)d_in[6];
  float* out = (float*)d_out;

  char* ws = (char*)d_ws;
  auto take = [&](size_t bytes) {
    void* p = (void*)ws;
    ws += (bytes + 255) & ~(size_t)255;
    return p;
  };
  const size_t M1 = (size_t)Bc * Sc;  // 4096
  __bf16* X   = (__bf16*)take(M1 * 3 * Dc * 2);
  __bf16* Wq  = (__bf16*)take((size_t)3 * Dc * 3 * Dc * 2);
  __bf16* Wob = (__bf16*)take((size_t)Dc * Dc * 2);
  __bf16* Qh  = (__bf16*)take((size_t)Bc * Hc * Sc * HDc * 2);
  __bf16* Kh  = (__bf16*)take((size_t)Bc * Hc * Sc * HDc * 2);
  __bf16* Vt  = (__bf16*)take((size_t)Bc * Hc * Sc * HDc * 2);
  __bf16* Ab  = (__bf16*)take((size_t)Bc * Sc * Dc * 2);

  {
    const size_t total = M1 * 3 * Dc;
    pack_x_kernel<<<(unsigned)((total + 255) / 256), 256, 0, stream>>>(q, kk, vv, X);
  }
  {
    const int n = 3 * Dc * 3 * Dc;
    cvt_kernel<<<(n + 255) / 256, 256, 0, stream>>>(Wqkv, Wq, n);
  }
  {
    const int n = Dc * Dc;
    cvt_kernel<<<(n + 255) / 256, 256, 0, stream>>>(Wo, Wob, n);
  }
  // QKV projection -> per-head Q/K (row) and V (transposed) bf16 buffers
  gemm_kernel<<<dim3(3 * Dc / 128, (unsigned)(M1 / 128)), 128, 0, stream>>>(
      X, Wq, bqkv, (int)M1, 3 * Dc, 3 * Dc, 0, Qh, Kh, Vt, nullptr);
  // flash attention
  attn_kernel<<<dim3(Sc / 64, Bc * Hc), 128, 0, stream>>>(Qh, Kh, Vt, Ab);
  // output projection -> fp32 d_out
  gemm_kernel<<<dim3(Dc / 128, (unsigned)(M1 / 128)), 128, 0, stream>>>(
      Ab, Wob, bo, (int)M1, Dc, Dc, 1, nullptr, nullptr, nullptr, out);
}
